// LTCcell_Model_29351806501440
// MI455X (gfx1250) — matrix operand order repaired and checked by reading, not compiled, not run on hardware
//
#include <hip/hip_runtime.h>
#include <stdint.h>

// ===========================================================================
// MI455X (gfx1250) LTC(track) + LTC(vertex) + MLP classifier.
//
// Roofline: LTC part = ~1e10 sigmoid evals (v_exp_f32 + v_rcp_f32 TRANS ops);
// params are LDS resident, HBM traffic ~25MB (~1us @23.3TB/s) -> pure
// VALU/TRANS bound. Thread==unit (wave32-clean), 2 batch elements per thread
// so each LDS parameter fetch feeds two sigmoids (keeps LDS well under the
// 64 dword/clk WGP supply while TRANS saturates). MLP part = 1.1 GFLOP GEMM
// -> v_wmma_f32_16x16x32_f16 with documented fragment layouts; weights
// pre-converted to transposed f16 once per call.
// ===========================================================================

#define BATCH 8192
#define L2E 1.4426950408889634f

typedef _Float16 h2   __attribute__((ext_vector_type(2)));
typedef _Float16 h8   __attribute__((ext_vector_type(8)));
typedef _Float16 v16h __attribute__((ext_vector_type(16)));
typedef float    v8f  __attribute__((ext_vector_type(8)));

#if defined(__HIP_DEVICE_COMPILE__) && __has_builtin(__builtin_amdgcn_exp2f)
#define EXP2F(x) __builtin_amdgcn_exp2f(x)
#else
#define EXP2F(x) exp2f(x)
#endif
#if defined(__HIP_DEVICE_COMPILE__) && __has_builtin(__builtin_amdgcn_rcpf)
#define RCPF(x) __builtin_amdgcn_rcpf(x)
#else
#define RCPF(x) (1.0f / (x))
#endif

// ---------------------------------------------------------------------------
// Fold LTC params:  sigmoid(sigma*(v-mu)) = rcp(1 + exp2(A*v + B))
//   AB.x = -sigma*log2(e), AB.y = sigma*mu*log2(e) (one ds_load_b64 in the
// hot loop).  Sensory version folds the input affine (i = x*iw+ib) into AB.
// w*mask and w*mask*erev packed as f16 pair (PK.x = den, PK.y = num weight).
// ---------------------------------------------------------------------------
__global__ void prep_ltc(const float* __restrict__ w, const float* __restrict__ sigma,
                         const float* __restrict__ mu, const float* __restrict__ erev,
                         const int* __restrict__ mask,
                         const float* __restrict__ s_w, const float* __restrict__ s_sigma,
                         const float* __restrict__ s_mu, const float* __restrict__ s_erev,
                         const int* __restrict__ s_mask,
                         const float* __restrict__ iw, const float* __restrict__ ib,
                         const float* __restrict__ gleak, const float* __restrict__ vleak,
                         const float* __restrict__ cm, int F,
                         float2* __restrict__ AB, h2* __restrict__ PK,
                         float2* __restrict__ sAB, h2* __restrict__ sPK,
                         float* __restrict__ vec) {
  int i = blockIdx.x * blockDim.x + threadIdx.x;
  if (i < 64 * 64) {
    float sg = sigma[i];
    float wm = w[i] * (float)mask[i];
    AB[i] = make_float2(-sg * L2E, sg * mu[i] * L2E);
    h2 p; p.x = (_Float16)wm; p.y = (_Float16)(wm * erev[i]);
    PK[i] = p;
  }
  if (i < F * 64) {
    int f = i >> 6;
    float sg = s_sigma[i];
    float a = -sg * L2E;
    float wm = s_w[i] * (float)s_mask[i];
    sAB[i] = make_float2(a * iw[f], sg * s_mu[i] * L2E + a * ib[f]);
    h2 p; p.x = (_Float16)wm; p.y = (_Float16)(wm * s_erev[i]);
    sPK[i] = p;
  }
  if (i < 64) {
    float g = gleak[i];
    vec[i] = cm[i] * 6.0f;            // cm * (ODE_UNFOLDS / ELAPSED)
    vec[64 + i] = g;
    vec[128 + i] = g * vleak[i];      // gleak*vleak
  }
}

// ---------------------------------------------------------------------------
// LTC sequence: 256-thread block = 8 batch elements; thread (tid&63) = unit,
// handling 2 batch elements so each parameter fetch feeds two sigmoids.
// Inner loop per j: ds_load_b64(AB) + ds_load_b32(PK) + 2 broadcast reads,
// then 2x { fma -> v_exp_f32 -> add -> v_rcp_f32 -> 2 fma }.
// Final state written straight into the f16 MLP input z at column colOff.
// ---------------------------------------------------------------------------
template <int F, int T>
__global__ void __launch_bounds__(256) ltc_forward(
    const float* __restrict__ x, const float2* __restrict__ gAB,
    const h2* __restrict__ gPK, const float2* __restrict__ gsAB,
    const h2* __restrict__ gsPK, const float* __restrict__ vec,
    _Float16* __restrict__ zout, int colOff) {
  __shared__ float2 lAB[64 * 64];    // 32 KB
  __shared__ h2     lPK[64 * 64];    // 16 KB
  __shared__ float2 lsAB[F * 64];
  __shared__ h2     lsPK[F * 64];
  __shared__ float  vsh[8][64];
  __shared__ float  xsh[8][F];

  const int tid = threadIdx.x;
  for (int i = tid; i < 64 * 64; i += 256) { lAB[i] = gAB[i]; lPK[i] = gPK[i]; }
  for (int i = tid; i < F * 64; i += 256) { lsAB[i] = gsAB[i]; lsPK[i] = gsPK[i]; }
  const int sub = tid >> 6;          // 0..3 -> pair of batch elements
  const int k   = tid & 63;          // unit index
  const int s0  = sub * 2, s1 = s0 + 1;
  const int b0  = blockIdx.x * 8 + s0;
  const float cm_t = vec[k];
  const float gl   = vec[64 + k];
  const float glvl = vec[128 + k];
  float v0 = 0.0f, v1 = 0.0f;
  vsh[s0][k] = 0.0f;
  vsh[s1][k] = 0.0f;
  __syncthreads();

  const float* xb0 = x + (size_t)b0 * T * F;
  const float* xb1 = xb0 + (size_t)T * F;
  for (int tt = T - 1; tt >= 0; --tt) {        // reference scans x[:, ::-1]
    if (k < F) {
      xsh[s0][k] = xb0[tt * F + k];
      xsh[s1][k] = xb1[tt * F + k];
    }
    __syncthreads();
    float sn0 = 0.0f, sd0 = 0.0f, sn1 = 0.0f, sd1 = 0.0f;
#pragma unroll
    for (int f = 0; f < F; ++f) {
      float2 ab = lsAB[f * 64 + k];
      h2 p = lsPK[f * 64 + k];
      float px = (float)p.x, py = (float)p.y;
      float e0 = EXP2F(fmaf(ab.x, xsh[s0][f], ab.y));
      float e1 = EXP2F(fmaf(ab.x, xsh[s1][f], ab.y));
      float q0 = RCPF(1.0f + e0);
      float q1 = RCPF(1.0f + e1);
      sn0 = fmaf(q0, py, sn0); sd0 = fmaf(q0, px, sd0);
      sn1 = fmaf(q1, py, sn1); sd1 = fmaf(q1, px, sd1);
    }
#pragma unroll 1
    for (int u = 0; u < 6; ++u) {
      float n0 = sn0, d0 = sd0, n1 = sn1, d1 = sd1;
#pragma unroll 4
      for (int j = 0; j < 64; ++j) {
        float2 ab = lAB[j * 64 + k];
        h2 p = lPK[j * 64 + k];
        float px = (float)p.x, py = (float)p.y;
        float e0 = EXP2F(fmaf(ab.x, vsh[s0][j], ab.y));
        float e1 = EXP2F(fmaf(ab.x, vsh[s1][j], ab.y));
        float q0 = RCPF(1.0f + e0);
        float q1 = RCPF(1.0f + e1);
        n0 = fmaf(q0, py, n0); d0 = fmaf(q0, px, d0);
        n1 = fmaf(q1, py, n1); d1 = fmaf(q1, px, d1);
      }
      v0 = (fmaf(cm_t, v0, glvl) + n0) * RCPF(cm_t + gl + d0 + 1e-8f);
      v1 = (fmaf(cm_t, v1, glvl) + n1) * RCPF(cm_t + gl + d1 + 1e-8f);
      __syncthreads();
      vsh[s0][k] = v0;
      vsh[s1][k] = v1;
      __syncthreads();
    }
  }
  zout[(size_t)b0 * 160 + colOff + k]       = (_Float16)v0;
  zout[(size_t)(b0 + 1) * 160 + colOff + k] = (_Float16)v1;
}

// ---------------------------------------------------------------------------
// Convert MLP weights to transposed f16: W?t[n][k] = w?[k][n] (K padded
// 134->160 with zeros for layer 1) so WMMA B-fragments are two contiguous 16B loads.
// ---------------------------------------------------------------------------
__global__ void prep_mlp(const float* __restrict__ w1, const float* __restrict__ w2,
                         const float* __restrict__ w3, const float* __restrict__ w4,
                         const float* __restrict__ wf,
                         _Float16* __restrict__ W1t, _Float16* __restrict__ W2t,
                         _Float16* __restrict__ W3t, _Float16* __restrict__ W4t,
                         _Float16* __restrict__ wf16) {
  int i = blockIdx.x * blockDim.x + threadIdx.x;
  if (i < 128 * 160) {
    int n = i / 160, kk = i % 160;
    W1t[i] = (kk < 134) ? (_Float16)w1[kk * 128 + n] : (_Float16)0.0f;
  }
  if (i < 128 * 128) {
    int n = i >> 7, kk = i & 127;
    W2t[i] = (_Float16)w2[kk * 128 + n];
    W3t[i] = (_Float16)w3[kk * 128 + n];
    W4t[i] = (_Float16)w4[kk * 128 + n];
  }
  if (i < 128) wf16[i] = (_Float16)wf[i];
}

// Fill z columns 128..159 with [jet | 0-pad]; LTC kernels own cols 0..127.
__global__ void pack_jet(const float* __restrict__ jet, _Float16* __restrict__ z) {
  int i = blockIdx.x * blockDim.x + threadIdx.x;
  if (i >= BATCH * 32) return;
  int b = i >> 5, c = i & 31;
  float v = (c < 6) ? jet[b * 6 + c] : 0.0f;
  z[(size_t)b * 160 + 128 + c] = (_Float16)v;
}

// ---------------------------------------------------------------------------
// WMMA MLP: block = 256 thr (8 wave32) = 16 batch rows; wave w owns cols
// 16w..16w+15.  K chunks of 32 via v_wmma_f32_16x16x32_f16.
// 16-bit A 16x32 layout: lanes 0-15 (M=lane): halves 0-7 -> K 0-7, 8-15 ->
// K 16-23; lanes 16-31: K 8-15 / 24-31  => two contiguous 16B loads per frag.
// B 32x16 layout: N = lane%16, K same as A (two 16B loads per frag).
// Hidden activations relu'd to f16 LDS (row stride 136 halves, bank-clean).
// ---------------------------------------------------------------------------
#define HPAD 136

__device__ __forceinline__ v16h frag16(const _Float16* p0, const _Float16* p1) {
  h8 lo = *(const h8*)p0;
  h8 hi = *(const h8*)p1;
  return __builtin_shufflevector(lo, hi, 0, 1, 2, 3, 4, 5, 6, 7,
                                 8, 9, 10, 11, 12, 13, 14, 15);
}

__global__ void __launch_bounds__(256) mlp_forward(
    const _Float16* __restrict__ z, const _Float16* __restrict__ W1t,
    const _Float16* __restrict__ W2t, const _Float16* __restrict__ W3t,
    const _Float16* __restrict__ W4t, const _Float16* __restrict__ wf16,
    const float* __restrict__ b1, const float* __restrict__ b2,
    const float* __restrict__ b3, const float* __restrict__ b4,
    const float* __restrict__ bf, float* __restrict__ out) {
  __shared__ __align__(16) _Float16 hbuf[2][16 * HPAD];
  const int tid  = threadIdx.x;
  const int lane = tid & 31;
  const int wv   = tid >> 5;       // wave id = output col tile
  const int rowM = lane & 15;
  const int hi16 = lane >> 4;
  const int col  = wv * 16 + rowM;
  const int row0 = blockIdx.x * 16;

  // ---- layer 1: A from global z [16 rows x 160], K = 5 chunks
  {
    v8f acc = {};
    const _Float16* ar = z + (size_t)(row0 + rowM) * 160;
    const _Float16* br = W1t + (size_t)col * 160;
#pragma unroll
    for (int kc = 0; kc < 5; ++kc) {
      v16h a  = frag16(ar + kc * 32 + hi16 * 8,  ar + kc * 32 + hi16 * 8 + 16);
      v16h bm = frag16(br + kc * 32 + hi16 * 8,  br + kc * 32 + hi16 * 8 + 16);
      acc = __builtin_amdgcn_wmma_f32_16x16x32_f16(false, a, false, bm,
                                                   (short)0, acc, false, false);
    }
    float bias = b1[col];
#pragma unroll
    for (int r = 0; r < 8; ++r) {   // C layout: M = r + 8*hi16, N = lane%16
      float vv = acc[r] + bias;
      hbuf[0][(r + hi16 * 8) * HPAD + col] = (_Float16)(vv > 0.0f ? vv : 0.0f);
    }
  }
  __syncthreads();

  // ---- layers 2..4: A from LDS, K = 4 chunks of 32
  const _Float16* Wts[3] = {W2t, W3t, W4t};
  const float*    bbs[3] = {b2, b3, b4};
  int cur = 0;
#pragma unroll
  for (int L = 0; L < 3; ++L) {
    v8f acc = {};
    const _Float16* ar = &hbuf[cur][rowM * HPAD];
    const _Float16* br = Wts[L] + (size_t)col * 128;
#pragma unroll
    for (int kc = 0; kc < 4; ++kc) {
      v16h a  = frag16(ar + kc * 32 + hi16 * 8,  ar + kc * 32 + hi16 * 8 + 16);
      v16h bm = frag16(br + kc * 32 + hi16 * 8,  br + kc * 32 + hi16 * 8 + 16);
      acc = __builtin_amdgcn_wmma_f32_16x16x32_f16(false, a, false, bm,
                                                   (short)0, acc, false, false);
    }
    float bias = bbs[L][col];
    __syncthreads();
#pragma unroll
    for (int r = 0; r < 8; ++r) {
      float vv = acc[r] + bias;
      hbuf[1 - cur][(r + hi16 * 8) * HPAD + col] = (_Float16)(vv > 0.0f ? vv : 0.0f);
    }
    __syncthreads();
    cur = 1 - cur;
  }

  // ---- final 128->1 + sigmoid (cheap, VALU)
  if (tid < 16) {
    const _Float16* hr = &hbuf[cur][tid * HPAD];
    float a2 = bf[0];
#pragma unroll 8
    for (int kk = 0; kk < 128; ++kk)
      a2 = fmaf((float)hr[kk], (float)wf16[kk], a2);
    out[row0 + tid] = RCPF(1.0f + EXP2F(-L2E * a2));
  }
}

// ---------------------------------------------------------------------------
// Input index map (top-level dict order; nested dicts flattened in jax
// pytree sorted-key order):
//  0 track_x  1 vertex_x  2 jet_x
//  track params  @3 : cm,erev,gleak,input_b,input_w,mu,s_erev,s_mu,s_sigma,
//                     s_w,sigma,vleak,w
//  vertex params @16: same order
//  mlp params    @29: b1,b2,b3,b4,bf,w1,w2,w3,w4,wf
//  masks         @39: track_mask, track_s_mask, vertex_mask, vertex_s_mask
// ---------------------------------------------------------------------------
extern "C" void kernel_launch(void* const* d_in, const int* in_sizes, int n_in,
                              void* d_out, int out_size, void* d_ws, size_t ws_size,
                              hipStream_t stream) {
  (void)in_sizes; (void)n_in; (void)out_size; (void)ws_size;
  auto F32 = [&](int i) { return (const float*)d_in[i]; };
  auto I32 = [&](int i) { return (const int*)d_in[i]; };

  uintptr_t ws = (uintptr_t)d_ws;
  auto alloc = [&](size_t bytes) -> void* {
    ws = (ws + 255) & ~(uintptr_t)255;
    void* p = (void*)ws;
    ws += bytes;
    return p;
  };

  // track derived params
  float2* tAB  = (float2*)alloc(64 * 64 * 8);
  h2*     tPK  = (h2*)    alloc(64 * 64 * 4);
  float2* tsAB = (float2*)alloc(16 * 64 * 8);
  h2*     tsPK = (h2*)    alloc(16 * 64 * 4);
  float*  tvec = (float*) alloc(3 * 64 * 4);
  // vertex derived params
  float2* vAB  = (float2*)alloc(64 * 64 * 8);
  h2*     vPK  = (h2*)    alloc(64 * 64 * 4);
  float2* vsAB = (float2*)alloc(12 * 64 * 8);
  h2*     vsPK = (h2*)    alloc(12 * 64 * 4);
  float*  vvec = (float*) alloc(3 * 64 * 4);
  // packed f16 MLP input + f16 weights
  _Float16* zf   = (_Float16*)alloc((size_t)BATCH * 160 * 2);
  _Float16* W1t  = (_Float16*)alloc(128 * 160 * 2);
  _Float16* W2t  = (_Float16*)alloc(128 * 128 * 2);
  _Float16* W3t  = (_Float16*)alloc(128 * 128 * 2);
  _Float16* W4t  = (_Float16*)alloc(128 * 128 * 2);
  _Float16* wf16 = (_Float16*)alloc(128 * 2);

  // param folding (track: F=16 / vertex: F=12)
  prep_ltc<<<16, 256, 0, stream>>>(F32(15), F32(13), F32(8), F32(4), I32(39),
                                   F32(12), F32(11), F32(10), F32(9), I32(40),
                                   F32(7), F32(6), F32(5), F32(14), F32(3), 16,
                                   tAB, tPK, tsAB, tsPK, tvec);
  prep_ltc<<<16, 256, 0, stream>>>(F32(28), F32(26), F32(21), F32(17), I32(41),
                                   F32(25), F32(24), F32(23), F32(22), I32(42),
                                   F32(20), F32(19), F32(18), F32(27), F32(16), 12,
                                   vAB, vPK, vsAB, vsPK, vvec);
  prep_mlp<<<80, 256, 0, stream>>>(F32(34), F32(35), F32(36), F32(37), F32(38),
                                   W1t, W2t, W3t, W4t, wf16);

  // LTC scans (the transcendental-bound bulk of the work); write z directly
  ltc_forward<16, 40><<<BATCH / 8, 256, 0, stream>>>(F32(0), tAB, tPK, tsAB,
                                                     tsPK, tvec, zf, 0);
  ltc_forward<12, 10><<<BATCH / 8, 256, 0, stream>>>(F32(1), vAB, vPK, vsAB,
                                                     vsPK, vvec, zf, 64);
  pack_jet<<<(BATCH * 32) / 256, 256, 0, stream>>>(F32(2), zf);

  // MLP head (WMMA)
  mlp_forward<<<BATCH / 16, 256, 0, stream>>>(zf, W1t, W2t, W3t, W4t, wf16,
                                              F32(29), F32(30), F32(31), F32(32),
                                              F32(33), (float*)d_out);
}
